// ReckoningLayer_44178033607295
// MI455X (gfx1250) — matrix operand order repaired and checked by reading, not compiled, not run on hardware
//
#include <hip/hip_runtime.h>
#include <hip/hip_bf16.h>

// ---------------------------------------------------------------------------
// ReckoningLayer on MI455X (gfx1250): bf16 WMMA GEMMs + fused epilogues.
// Round 3: GEMM templated on K so all fragment addresses fold into 24-bit
// instruction immediates (no per-iteration 64-bit address VALU), unroll-2
// K loop for deeper load/WMMA overlap. Chunked 3-pass scan unchanged.
// ---------------------------------------------------------------------------

typedef __attribute__((ext_vector_type(16))) __bf16 v16bf;
typedef __attribute__((ext_vector_type(8)))  __bf16 v8bf;
typedef __attribute__((ext_vector_type(8)))  float  v8f;

constexpr int BB = 4;
constexpr int TT = 2048;
constexpr int DD = 1024;
constexpr int DMM = 256;
constexpr int KC = 7;
constexpr int DFF = 4096;
constexpr int BT = BB * TT;      // 8192 tokens
constexpr int D3 = 3 * DD;       // 3072
constexpr int NCH = 32;          // scan chunks per sequence
constexpr int CL = TT / NCH;     // 64 steps per chunk

__device__ __forceinline__ float sigmoidf_dev(float x) {
  return 1.0f / (1.0f + __expf(-x));
}

// ---------------------------------------------------------------------------
// Weight convert (+ optional transpose): Bt[n*K + k] = W[k*N + n], bf16.
// ---------------------------------------------------------------------------
__global__ __launch_bounds__(256) void convert_tr_kernel(
    const float* __restrict__ W, __bf16* __restrict__ Wt,
    int R, int C, int transpose) {
  int idx = blockIdx.x * 256 + threadIdx.x;
  if (idx >= R * C) return;
  int r = idx / C, c = idx % C;
  if (transpose)
    Wt[(size_t)c * R + r] = (__bf16)W[idx];
  else
    Wt[idx] = (__bf16)W[idx];
}

// ---------------------------------------------------------------------------
// RMSNorm over rows of length Ddim; optional fp32 out + bf16 out.
// ---------------------------------------------------------------------------
__global__ __launch_bounds__(256) void rmsnorm_kernel(
    const float* __restrict__ x, const float* __restrict__ w,
    float* __restrict__ hf, __bf16* __restrict__ hb, int Ddim) {
  __shared__ float red[256];
  size_t row = blockIdx.x;
  const float* xr = x + row * Ddim;
  float ss = 0.0f;
  for (int i = threadIdx.x; i < Ddim; i += 256) {
    float v = xr[i];
    ss += v * v;
  }
  red[threadIdx.x] = ss;
  __syncthreads();
  for (int s = 128; s > 0; s >>= 1) {
    if (threadIdx.x < (unsigned)s) red[threadIdx.x] += red[threadIdx.x + s];
    __syncthreads();
  }
  float r = rsqrtf(red[0] / (float)Ddim + 1e-6f);
  for (int i = threadIdx.x; i < Ddim; i += 256) {
    float v = xr[i] * r * w[i];
    if (hf) hf[row * Ddim + i] = v;
    hb[row * Ddim + i] = (__bf16)v;
  }
}

// ---------------------------------------------------------------------------
// Causal depthwise conv K=7 over h[BT,D]; writes bf16 into cat slice 0
// (cat row stride = 3*D).
// ---------------------------------------------------------------------------
__global__ __launch_bounds__(256) void conv_kernel(
    const float* __restrict__ h, const float* __restrict__ cw,
    __bf16* __restrict__ cat) {
  size_t idx = (size_t)blockIdx.x * 256 + threadIdx.x;  // over BT*D
  int d = (int)(idx % DD);
  size_t row = idx / DD;  // b*T + t
  int t = (int)(row % TT);
  float acc = 0.0f;
#pragma unroll
  for (int j = 0; j < KC; j++) {
    int tt = t + j - (KC - 1);
    if (tt >= 0) acc += h[(row + (size_t)(j - (KC - 1))) * DD + d] * cw[d * KC + j];
  }
  cat[row * D3 + d] = (__bf16)acc;
}

// ---------------------------------------------------------------------------
// Chunked parallel scan for s_t = a_t * s_{t-1} + u_t.
// Pass 1: per (chunk, b, d): P = prod(a), Q = chunk-local scan end (s_in = 0).
// Pass 2: per (b, d): serial combine of 32 chunk summaries -> incoming carry.
// Pass 3: per (chunk, b, d): replay chunk from carry, write bf16 cat slice 1.
// ---------------------------------------------------------------------------
__global__ __launch_bounds__(256) void scan_pass1(
    const float* __restrict__ decay, const float* __restrict__ upd,
    float* __restrict__ P, float* __restrict__ Q) {
  int tid = blockIdx.x * 256 + threadIdx.x;  // over NCH * B * D
  if (tid >= NCH * BB * DD) return;
  int bd = tid % (BB * DD);
  int c = tid / (BB * DD);
  int d = bd % DD;
  int b = bd / DD;
  size_t base = ((size_t)b * TT + (size_t)c * CL) * DD + d;
  float p = 1.0f, q = 0.0f;
  for (int t = 0; t < CL; t++) {
    float a = decay[base + (size_t)t * DD];
    float u = upd[base + (size_t)t * DD];
    p *= a;
    q = a * q + u;
  }
  P[tid] = p;
  Q[tid] = q;
}

__global__ __launch_bounds__(256) void scan_pass2(
    const float* __restrict__ P, const float* __restrict__ Q,
    float* __restrict__ carry) {
  int bd = blockIdx.x * 256 + threadIdx.x;  // over B * D
  if (bd >= BB * DD) return;
  float s = 0.0f;
#pragma unroll
  for (int c = 0; c < NCH; c++) {
    int ix = c * (BB * DD) + bd;
    carry[ix] = s;
    s = P[ix] * s + Q[ix];
  }
}

__global__ __launch_bounds__(256) void scan_pass3(
    const float* __restrict__ decay, const float* __restrict__ upd,
    const float* __restrict__ carry, const float* __restrict__ scale_ptr,
    __bf16* __restrict__ cat) {
  int tid = blockIdx.x * 256 + threadIdx.x;  // over NCH * B * D
  if (tid >= NCH * BB * DD) return;
  int bd = tid % (BB * DD);
  int c = tid / (BB * DD);
  int d = bd % DD;
  int b = bd / DD;
  float sc = *scale_ptr;
  float s = carry[tid];
  size_t rowbase = (size_t)b * TT + (size_t)c * CL;
  for (int t = 0; t < CL; t++) {
    size_t row = rowbase + t;
    size_t ix = row * DD + d;
    s = decay[ix] * s + upd[ix];
    cat[row * D3 + DD + d] = (__bf16)(s * sc);
  }
}

// ---------------------------------------------------------------------------
// bf16 WMMA GEMM: C[M,N] = A[M,K] * Bt[N,K]^T, fp32 accumulate, K compile-time.
// Block: 256 threads (8 waves) arranged 4(M) x 2(N); block tile 128 x 128.
// Wave tile 32 x 64 = 2x4 WMMA 16x16 tiles; K stepped by 32, unrolled x2.
// All fragment addresses = single base + compile-time immediate + k.
// Epilogues:
//   0: Cb = acc * (*scale or 1)                       (bf16 out, ldc stride)
//   1: Cf = clip(sigmoid(acc + bias[n]), .05, .95)    (fp32)
//   2: Cf = sigmoid(acc + bias[n]) * auxf[m,n]        (fp32)   [updates]
//   3: Cf = acc + bias[n] + auxf[m,n]                 (fp32)   [mixed]
//   5: Cb = silu(auxb[m,n]) * acc                     (bf16)   [SwiGLU]
//   6: Cf = acc + auxf[m,n]                           (fp32)   [out=mixed+ff]
// ---------------------------------------------------------------------------
template <int EP, int K>
__global__ __launch_bounds__(256) void gemm_bf16_ep(
    const __bf16* __restrict__ A, const __bf16* __restrict__ Bt,
    float* __restrict__ Cf, __bf16* __restrict__ Cb, int ldc,
    const float* __restrict__ bias,
    const float* __restrict__ auxf, const __bf16* __restrict__ auxb,
    const float* __restrict__ scale_ptr) {
  const int lane = threadIdx.x & 31;
  const int wave = threadIdx.x >> 5;  // 0..7
  const int wm = wave >> 1;           // 0..3 -> M (32 rows each)
  const int wn = wave & 1;            // 0..1 -> N (64 cols each)
  const int bM = blockIdx.x * 128;
  const int bN = blockIdx.y * 128;

  const int m16 = lane & 15;
  const int hi = lane >> 4;
  const int kloA = hi * 16;   // A: elems 0..7 = K+kloA..,  8..15 = K+8+kloA.. (contig)
  const int kloB = hi * 16;   // B: elems 0..15 = K+kloB..K+kloB+15 (contig)

  // Single bases; all other offsets are compile-time constants.
  const __bf16* Abase = A + (size_t)(bM + wm * 32 + m16) * K + kloA;
  const __bf16* Bbase = Bt + (size_t)(bN + wn * 64 + m16) * K + kloB;

  v8f acc[2][4] = {};

#pragma unroll 2
  for (int k = 0; k < K; k += 32) {
    v8bf a0lo = *(const v8bf*)(Abase + k);
    v8bf a0hi = *(const v8bf*)(Abase + k + 8);
    v8bf a1lo = *(const v8bf*)(Abase + 16 * K + k);
    v8bf a1hi = *(const v8bf*)(Abase + 16 * K + k + 8);
    v16bf a0, a1;
#pragma unroll
    for (int i = 0; i < 8; i++) {
      a0[i] = a0lo[i]; a0[8 + i] = a0hi[i];
      a1[i] = a1lo[i]; a1[8 + i] = a1hi[i];
    }
    v16bf b[4];
#pragma unroll
    for (int ni = 0; ni < 4; ni++)
      b[ni] = *(const v16bf*)(Bbase + ni * 16 * K + k);

#pragma unroll
    for (int ni = 0; ni < 4; ni++) {
      acc[0][ni] = __builtin_amdgcn_wmma_f32_16x16x32_bf16(
          false, a0, false, b[ni], (short)0, acc[0][ni], false, false);
      acc[1][ni] = __builtin_amdgcn_wmma_f32_16x16x32_bf16(
          false, a1, false, b[ni], (short)0, acc[1][ni], false, false);
    }
  }

  float sc = 1.0f;
  if (EP == 0) sc = scale_ptr ? *scale_ptr : 1.0f;

#pragma unroll
  for (int mi = 0; mi < 2; mi++) {
#pragma unroll
    for (int ni = 0; ni < 4; ni++) {
      int rbase = bM + wm * 32 + mi * 16 + hi * 8;  // C: VGPR r -> M = hi*8+r
      int col = bN + wn * 64 + ni * 16 + m16;       //      N = lane&15
#pragma unroll
      for (int r = 0; r < 8; r++) {
        size_t mrow = (size_t)(rbase + r);
        size_t cix = mrow * ldc + col;
        float v = acc[mi][ni][r];
        if (EP == 0) {
          Cb[cix] = (__bf16)(v * sc);
        } else if (EP == 1) {
          float g = sigmoidf_dev(v + bias[col]);
          g = fminf(fmaxf(g, 0.05f), 0.95f);
          Cf[cix] = g;
        } else if (EP == 2) {
          float g = sigmoidf_dev(v + bias[col]);
          Cf[cix] = g * auxf[cix];
        } else if (EP == 3) {
          Cf[cix] = v + bias[col] + auxf[cix];
        } else if (EP == 5) {
          float g = (float)auxb[cix];
          Cb[cix] = (__bf16)((g * sigmoidf_dev(g)) * v);
        } else if (EP == 6) {
          Cf[cix] = v + auxf[cix];
        }
      }
    }
  }
}

// ---------------------------------------------------------------------------
// Host-side launcher.
// ---------------------------------------------------------------------------
static inline dim3 gemm_grid(int M, int N) { return dim3(M / 128, N / 128); }

extern "C" void kernel_launch(void* const* d_in, const int* in_sizes, int n_in,
                              void* d_out, int out_size, void* d_ws, size_t ws_size,
                              hipStream_t stream) {
  (void)in_sizes; (void)n_in; (void)out_size; (void)ws_size;

  const float* x        = (const float*)d_in[0];
  const float* conv_w   = (const float*)d_in[1];
  const float* norm1_w  = (const float*)d_in[2];
  const float* norm2_w  = (const float*)d_in[3];
  const float* Wdecay   = (const float*)d_in[4];
  const float* bdecay   = (const float*)d_in[5];
  const float* Wgate    = (const float*)d_in[6];
  const float* bgate    = (const float*)d_in[7];
  const float* state_sc = (const float*)d_in[8];
  const float* Wk       = (const float*)d_in[9];
  const float* Mw       = (const float*)d_in[10];
  const float* Wmem     = (const float*)d_in[11];
  const float* mem_sc   = (const float*)d_in[12];
  const float* Wc       = (const float*)d_in[13];
  const float* bc       = (const float*)d_in[14];
  const float* Wup      = (const float*)d_in[15];
  const float* Wg       = (const float*)d_in[16];
  const float* Wdown    = (const float*)d_in[17];
  float* out = (float*)d_out;

  char* ws = (char*)d_ws;
  size_t o = 0;
  auto alloc = [&](size_t bytes) {
    size_t cur = o;
    o += (bytes + 255) & ~(size_t)255;
    return cur;
  };

  float*  h_f   = (float*) (ws + alloc((size_t)BT * DD * 4));
  __bf16* h_b   = (__bf16*)(ws + alloc((size_t)BT * DD * 2));
  float*  decay = (float*) (ws + alloc((size_t)BT * DD * 4));
  float*  upd   = (float*) (ws + alloc((size_t)BT * DD * 4));
  __bf16* cat   = (__bf16*)(ws + alloc((size_t)BT * D3 * 2));
  __bf16* kbuf  = (__bf16*)(ws + alloc((size_t)BT * DMM * 2));
  __bf16* mo    = (__bf16*)(ws + alloc((size_t)BT * DMM * 2));
  float*  mixed = (float*) (ws + alloc((size_t)BT * DD * 4));
  __bf16* h2_b  = (__bf16*)(ws + alloc((size_t)BT * DD * 2));
  __bf16* Gbuf  = (__bf16*)(ws + alloc((size_t)BT * DFF * 2));
  __bf16* Vbuf  = (__bf16*)(ws + alloc((size_t)BT * DFF * 2));
  float*  Pbuf  = (float*) (ws + alloc((size_t)NCH * BB * DD * 4));
  float*  Qbuf  = (float*) (ws + alloc((size_t)NCH * BB * DD * 4));
  float*  carry = (float*) (ws + alloc((size_t)NCH * BB * DD * 4));
  __bf16* WdT   = (__bf16*)(ws + alloc((size_t)DD * DD * 2));
  __bf16* WgT   = (__bf16*)(ws + alloc((size_t)DD * DD * 2));
  __bf16* WkT   = (__bf16*)(ws + alloc((size_t)DD * DMM * 2));
  __bf16* Mb    = (__bf16*)(ws + alloc((size_t)DMM * DMM * 2));
  __bf16* WmT   = (__bf16*)(ws + alloc((size_t)DMM * DD * 2));
  __bf16* WcT   = (__bf16*)(ws + alloc((size_t)D3 * DD * 2));
  __bf16* WupT  = (__bf16*)(ws + alloc((size_t)DD * DFF * 2));
  __bf16* WggT  = (__bf16*)(ws + alloc((size_t)DD * DFF * 2));
  __bf16* WdnT  = (__bf16*)(ws + alloc((size_t)DFF * DD * 2));

  auto cvt = [&](const float* W, __bf16* Wt, int R, int C, int tr) {
    int n = R * C;
    convert_tr_kernel<<<(n + 255) / 256, 256, 0, stream>>>(W, Wt, R, C, tr);
  };

  // Weight conversions (Bt layout: [N, K] row-major, bf16).
  cvt(Wdecay, WdT, DD, DD, 1);
  cvt(Wgate,  WgT, DD, DD, 1);
  cvt(Wk,     WkT, DD, DMM, 1);
  cvt(Mw,     Mb,  DMM, DMM, 0);   // Bt for k@M^T is M itself
  cvt(Wmem,   WmT, DMM, DD, 1);
  cvt(Wc,     WcT, D3, DD, 1);
  cvt(Wup,    WupT, DD, DFF, 1);
  cvt(Wg,     WggT, DD, DFF, 1);
  cvt(Wdown,  WdnT, DFF, DD, 1);

  // h = rmsnorm(x, norm1_w)  (fp32 + bf16)
  rmsnorm_kernel<<<BT, 256, 0, stream>>>(x, norm1_w, h_f, h_b, DD);

  // decay = clip(sigmoid(h@Wdecay + bdecay))
  gemm_bf16_ep<1, DD><<<gemm_grid(BT, DD), 256, 0, stream>>>(
      h_b, WdT, decay, nullptr, DD, bdecay, nullptr, nullptr, nullptr);
  // updates = sigmoid(h@Wgate + bgate) * h
  gemm_bf16_ep<2, DD><<<gemm_grid(BT, DD), 256, 0, stream>>>(
      h_b, WgT, upd, nullptr, DD, bgate, h_f, nullptr, nullptr);

  // local -> cat[:, 0:D]
  conv_kernel<<<(BT * DD) / 256, 256, 0, stream>>>(h_f, conv_w, cat);

  // state -> cat[:, D:2D]  (3-pass chunked scan)
  scan_pass1<<<(NCH * BB * DD + 255) / 256, 256, 0, stream>>>(decay, upd, Pbuf, Qbuf);
  scan_pass2<<<(BB * DD + 255) / 256, 256, 0, stream>>>(Pbuf, Qbuf, carry);
  scan_pass3<<<(NCH * BB * DD + 255) / 256, 256, 0, stream>>>(decay, upd, carry,
                                                              state_sc, cat);

  // k = h@Wk ; mem_out = k@M^T ; mem = mem_out@Wmem * mem_scale -> cat[:, 2D:3D]
  gemm_bf16_ep<0, DD><<<gemm_grid(BT, DMM), 256, 0, stream>>>(
      h_b, WkT, nullptr, kbuf, DMM, nullptr, nullptr, nullptr, nullptr);
  gemm_bf16_ep<0, DMM><<<gemm_grid(BT, DMM), 256, 0, stream>>>(
      kbuf, Mb, nullptr, mo, DMM, nullptr, nullptr, nullptr, nullptr);
  gemm_bf16_ep<0, DMM><<<gemm_grid(BT, DD), 256, 0, stream>>>(
      mo, WmT, nullptr, cat + 2 * DD, D3, nullptr, nullptr, nullptr, mem_sc);

  // mixed = x + cat@Wc + bc
  gemm_bf16_ep<3, D3><<<gemm_grid(BT, DD), 256, 0, stream>>>(
      cat, WcT, mixed, nullptr, DD, bc, x, nullptr, nullptr);

  // h2 = rmsnorm(mixed, norm2_w)  (bf16 only)
  rmsnorm_kernel<<<BT, 256, 0, stream>>>(mixed, norm2_w, nullptr, h2_b, DD);

  // G = h2@Wg ; V = silu(G) * (h2@Wup)
  gemm_bf16_ep<0, DD><<<gemm_grid(BT, DFF), 256, 0, stream>>>(
      h2_b, WggT, nullptr, Gbuf, DFF, nullptr, nullptr, nullptr, nullptr);
  gemm_bf16_ep<5, DD><<<gemm_grid(BT, DFF), 256, 0, stream>>>(
      h2_b, WupT, nullptr, Vbuf, DFF, nullptr, nullptr, Gbuf, nullptr);

  // out = mixed + V@Wdown
  gemm_bf16_ep<6, DFF><<<gemm_grid(BT, DD), 256, 0, stream>>>(
      Vbuf, WdnT, out, nullptr, DD, nullptr, mixed, nullptr, nullptr);
}
